// DecoderRNN_37615323578978
// MI455X (gfx1250) — compile-verified, matrix operand order checked by reading, not run on hardware
//
#include <hip/hip_runtime.h>
#include <math.h>

typedef __attribute__((ext_vector_type(16))) __bf16 bf16x16;
typedef __attribute__((ext_vector_type(8)))  __bf16 bf16x8;
typedef __attribute__((ext_vector_type(4)))  __bf16 bf16x4;
typedef __attribute__((ext_vector_type(8)))  float  f32x8;

constexpr int kV = 32000;
constexpr int kE = 512;
constexpr int kH = 1024;
constexpr int kB = 64;
constexpr int kT = 30;
constexpr int kFCBlocks = kV / 256;   // 125
constexpr int kLstmWGs  = kH / 64;    // 16

// ---------------- fragment loaders (ISA 16-bit A/B layout) ----------------
// lane L<16 : row = L,   K 0..7 then K 16..23
// lane L>=16: row = L-16, K 8..15 then K 24..31
__device__ inline bf16x16 load_frag_bf16(const __bf16* __restrict__ row, int kb, int lane) {
  const int hi = (lane >> 4) & 1;
  const bf16x8* p = (const bf16x8*)(row + kb * 32 + hi * 8);
  bf16x8 lo = p[0];        // K base+0..7   (16B)
  bf16x8 hg = p[2];        // K base+16..23 (16B)
  return __builtin_shufflevector(lo, hg, 0,1,2,3,4,5,6,7,8,9,10,11,12,13,14,15);
}

__device__ inline bf16x16 load_frag_f32(const float* __restrict__ row, int kb, int lane) {
  const int hi = (lane >> 4) & 1;
  const float4* p = (const float4*)(row + kb * 32 + hi * 8);
  float4 a = p[0], b = p[1], c = p[4], d = p[5];
  bf16x16 r;
  r[0]=(__bf16)a.x;  r[1]=(__bf16)a.y;  r[2]=(__bf16)a.z;  r[3]=(__bf16)a.w;
  r[4]=(__bf16)b.x;  r[5]=(__bf16)b.y;  r[6]=(__bf16)b.z;  r[7]=(__bf16)b.w;
  r[8]=(__bf16)c.x;  r[9]=(__bf16)c.y;  r[10]=(__bf16)c.z; r[11]=(__bf16)c.w;
  r[12]=(__bf16)d.x; r[13]=(__bf16)d.y; r[14]=(__bf16)d.z; r[15]=(__bf16)d.w;
  return r;
}

__device__ inline float sigmoidf_(float x) { return 1.f / (1.f + __expf(-x)); }

// ---------------- simple deterministic grid barrier (per-step counter) ----------------
__device__ inline void grid_barrier(int* ctr, int nwg) {
  __syncthreads();
  if (threadIdx.x == 0) {
    __threadfence();
    atomicAdd(ctr, 1);
    while (__hip_atomic_load(ctr, __ATOMIC_ACQUIRE, __HIP_MEMORY_SCOPE_AGENT) < nwg)
      __builtin_amdgcn_s_sleep(1);
  }
  __syncthreads();
  __threadfence();
}

// ---------------- one LSTM layer: gates GEMM (WMMA) + fused cell update ----------------
// 8 waves: mt2 = w>>2 -> rows [32*mt2,+32); g = w&3 -> gate. Cols = [hbase,hbase+64).
template<bool GATHER>
__device__ void lstm_layer(float (&gl)[4][64][65],
                           const float* __restrict__ embf, const int* __restrict__ tok,
                           const __bf16* __restrict__ a1, int Ka,
                           const __bf16* __restrict__ wih,
                           const __bf16* __restrict__ hprev,
                           const __bf16* __restrict__ whh,
                           const float* __restrict__ bcomb,
                           float* __restrict__ c,          // in-place cell state
                           __bf16* __restrict__ hnew,
                           int hbase, int tid)
{
  const int lane = tid & 31;
  const int wave = tid >> 5;
  const int mt2  = wave >> 2;
  const int g    = wave & 3;
  const int l15  = lane & 15;
  const int lhi  = (lane >> 4) & 1;

  f32x8 acc[2][4] = {};

  // Phase 1: X @ W_ih^T
  for (int kb = 0; kb < Ka / 32; ++kb) {
    bf16x16 af[2];
#pragma unroll
    for (int mi = 0; mi < 2; ++mi) {
      const int row = mt2 * 32 + mi * 16 + l15;
      if (GATHER) af[mi] = load_frag_f32(embf + (size_t)tok[row] * kE, kb, lane);
      else        af[mi] = load_frag_bf16(a1 + (size_t)row * Ka, kb, lane);
    }
    bf16x16 bfr[4];
#pragma unroll
    for (int nj = 0; nj < 4; ++nj)
      bfr[nj] = load_frag_bf16(wih + (size_t)(g * kH + hbase + nj * 16 + l15) * Ka, kb, lane);
#pragma unroll
    for (int mi = 0; mi < 2; ++mi)
#pragma unroll
      for (int nj = 0; nj < 4; ++nj)
        acc[mi][nj] = __builtin_amdgcn_wmma_f32_16x16x32_bf16(
            false, af[mi], false, bfr[nj], (short)0, acc[mi][nj], false, false);
  }
  // Phase 2: H_prev @ W_hh^T
  for (int kb = 0; kb < kH / 32; ++kb) {
    bf16x16 af[2];
#pragma unroll
    for (int mi = 0; mi < 2; ++mi)
      af[mi] = load_frag_bf16(hprev + (size_t)(mt2 * 32 + mi * 16 + l15) * kH, kb, lane);
    bf16x16 bfr[4];
#pragma unroll
    for (int nj = 0; nj < 4; ++nj)
      bfr[nj] = load_frag_bf16(whh + (size_t)(g * kH + hbase + nj * 16 + l15) * kH, kb, lane);
#pragma unroll
    for (int mi = 0; mi < 2; ++mi)
#pragma unroll
      for (int nj = 0; nj < 4; ++nj)
        acc[mi][nj] = __builtin_amdgcn_wmma_f32_16x16x32_bf16(
            false, af[mi], false, bfr[nj], (short)0, acc[mi][nj], false, false);
  }

  // Stage gate pre-activations (C/D layout: VGPR r -> M=r, hi lanes +8)
#pragma unroll
  for (int mi = 0; mi < 2; ++mi)
#pragma unroll
    for (int nj = 0; nj < 4; ++nj)
#pragma unroll
      for (int r = 0; r < 8; ++r)
        gl[g][mt2 * 32 + mi * 16 + lhi * 8 + r][nj * 16 + l15] = acc[mi][nj][r];
  __syncthreads();

  // Fused LSTM cell update
  {
    const int b = tid >> 2;
    const int q = tid & 3;
#pragma unroll 4
    for (int j = 0; j < 16; ++j) {
      const int col = q * 16 + j;
      const int hg  = hbase + col;
      const float gi = gl[0][b][col] + bcomb[0 * kH + hg];
      const float gf = gl[1][b][col] + bcomb[1 * kH + hg];
      const float gg = gl[2][b][col] + bcomb[2 * kH + hg];
      const float go = gl[3][b][col] + bcomb[3 * kH + hg];
      const float cc = sigmoidf_(gf) * c[(size_t)b * kH + hg] + sigmoidf_(gi) * tanhf(gg);
      const float hh = sigmoidf_(go) * tanhf(cc);
      c[(size_t)b * kH + hg] = cc;
      hnew[(size_t)b * kH + hg] = (__bf16)hh;
    }
  }
}

// ---------------- fused 2-layer LSTM step (grid barrier between layers) ----------------
__global__ __launch_bounds__(256)
void lstm2_step_kernel(const float* __restrict__ embf, const int* __restrict__ tok,
                       const __bf16* __restrict__ wih0, const __bf16* __restrict__ whh0,
                       const float* __restrict__ b0c,
                       const __bf16* __restrict__ wih1, const __bf16* __restrict__ whh1,
                       const float* __restrict__ b1c,
                       const __bf16* __restrict__ h0prev, __bf16* __restrict__ h0next,
                       float* __restrict__ c0,
                       const __bf16* __restrict__ h1prev, __bf16* __restrict__ h1next,
                       float* __restrict__ c1,
                       int* __restrict__ ctr)
{
  __shared__ float gl[4][64][65];
  const int tid = threadIdx.x;
  const int hbase = blockIdx.x * 64;

  lstm_layer<true>(gl, embf, tok, (const __bf16*)nullptr, kE,
                   wih0, h0prev, whh0, b0c, c0, h0next, hbase, tid);
  grid_barrier(ctr, kLstmWGs);   // all h0next visible before layer 1 consumes full H
  lstm_layer<false>(gl, (const float*)nullptr, (const int*)nullptr, h0next, kH,
                    wih1, h1prev, whh1, b1c, c1, h1next, hbase, tid);
}

// ---------------- FC logits GEMM (split-K) + softmax partials + fused tail reduce ----------------
// 8 waves: kw = w>>2 -> K half; nq = w&3 -> 64-col group. Each wave: 4M x 4N tiles.
// fc_w fragments are unique per WG (weights read exactly once per step).
__global__ __launch_bounds__(256)
void fc_logits_kernel(const __bf16* __restrict__ h1,     // [B, H] bf16
                      const __bf16* __restrict__ fcw,    // [V, H] bf16
                      const float* __restrict__ fcb,     // [V]
                      float* __restrict__ out,           // [B, T, V]
                      int t,
                      float* __restrict__ pm, int* __restrict__ pa, float* __restrict__ ps,
                      int* __restrict__ ctr,
                      int* __restrict__ tok, float* __restrict__ lse)
{
  __shared__ float lg[64][260];          // logits tile (rows stay 16B aligned)
  __shared__ float rm[64][4];
  __shared__ int   ra[64][4];
  __shared__ float rs[64][4];
  __shared__ int   sdone;

  const int tid  = threadIdx.x;
  const int lane = tid & 31;
  const int wave = tid >> 5;
  const int kw   = wave >> 2;            // K half: kb in [kw*16, kw*16+16)
  const int nq   = wave & 3;
  const int vbase = blockIdx.x * 256;
  const int l15  = lane & 15;
  const int lhi  = (lane >> 4) & 1;

  f32x8 acc[4][4] = {};

  for (int kb = kw * 16; kb < kw * 16 + 16; ++kb) {
    bf16x16 af[4];
#pragma unroll
    for (int mi = 0; mi < 4; ++mi)
      af[mi] = load_frag_bf16(h1 + (size_t)(mi * 16 + l15) * kH, kb, lane);
    bf16x16 bfr[4];
#pragma unroll
    for (int nj = 0; nj < 4; ++nj)
      bfr[nj] = load_frag_bf16(fcw + (size_t)(vbase + nq * 64 + nj * 16 + l15) * kH, kb, lane);
#pragma unroll
    for (int mi = 0; mi < 4; ++mi)
#pragma unroll
      for (int nj = 0; nj < 4; ++nj)
        acc[mi][nj] = __builtin_amdgcn_wmma_f32_16x16x32_bf16(
            false, af[mi], false, bfr[nj], (short)0, acc[mi][nj], false, false);
  }

  // combine K halves in LDS: kw==0 writes (+bias), then kw==1 accumulates
  if (kw == 0) {
#pragma unroll
    for (int nj = 0; nj < 4; ++nj) {
      const float fb = fcb[vbase + nq * 64 + nj * 16 + l15];
#pragma unroll
      for (int mi = 0; mi < 4; ++mi)
#pragma unroll
        for (int r = 0; r < 8; ++r)
          lg[mi * 16 + lhi * 8 + r][nq * 64 + nj * 16 + l15] = acc[mi][nj][r] + fb;
    }
  }
  __syncthreads();
  if (kw == 1) {
#pragma unroll
    for (int nj = 0; nj < 4; ++nj)
#pragma unroll
      for (int mi = 0; mi < 4; ++mi)
#pragma unroll
        for (int r = 0; r < 8; ++r)
          lg[mi * 16 + lhi * 8 + r][nq * 64 + nj * 16 + l15] += acc[mi][nj][r];
  }
  __syncthreads();

  // coalesced logit write-out: [B, T, V]
  for (int i = tid; i < 64 * 64; i += 256) {
    const int bb = i >> 6;
    const int q4 = i & 63;
    float4 v = *(const float4*)&lg[bb][q4 * 4];
    *(float4*)&out[((size_t)bb * kT + t) * kV + vbase + q4 * 4] = v;
  }

  // per-block online-softmax partials: 4 threads per row, 64 cols each
  {
    const int b = tid >> 2;
    const int q = tid & 3;
    float m = -INFINITY; int am = 0;
    for (int j = 0; j < 64; ++j) {
      const float v = lg[b][q * 64 + j];
      if (v > m) { m = v; am = vbase + q * 64 + j; }
    }
    rm[b][q] = m; ra[b][q] = am;
    __syncthreads();
    float bm = rm[b][0];
#pragma unroll
    for (int qq = 1; qq < 4; ++qq) bm = fmaxf(bm, rm[b][qq]);
    float s = 0.f;
    for (int j = 0; j < 64; ++j) s += __expf(lg[b][q * 64 + j] - bm);
    rs[b][q] = s;
    __syncthreads();
    if (q == 0) {
      float mm = -INFINITY; int aa = 0; float ss = 0.f;
#pragma unroll
      for (int qq = 0; qq < 4; ++qq) {
        if (rm[b][qq] > mm) { mm = rm[b][qq]; aa = ra[b][qq]; }
        ss += rs[b][qq];
      }
      pm[blockIdx.x * 64 + b] = mm;
      pa[blockIdx.x * 64 + b] = aa;
      ps[blockIdx.x * 64 + b] = ss;
    }
  }

  // fused tail: last-arriving block reduces across all 125 blocks
  __threadfence();
  __syncthreads();
  if (tid == 0) {
    const int old = atomicAdd(ctr, 1);
    sdone = (old == kFCBlocks - 1) ? 1 : 0;
  }
  __syncthreads();
  if (sdone) {
    __threadfence();                      // acquire all blocks' partials
    if (tid < kB) {
      const int b = tid;
      float m = -INFINITY; int arg = 0;
      for (int blk = 0; blk < kFCBlocks; ++blk) {
        const float v = pm[blk * 64 + b];
        if (v > m) { m = v; arg = pa[blk * 64 + b]; }   // strict '>' keeps first occurrence
      }
      float s = 0.f;
      for (int blk = 0; blk < kFCBlocks; ++blk)
        s += ps[blk * 64 + b] * __expf(pm[blk * 64 + b] - m);
      tok[b] = arg;                       // greedy feedback
      lse[b * kT + t] = m + __logf(s);
    }
  }
}

// ---------------- small utility kernels ----------------
__global__ void cvt_f32_bf16_kernel(const float* __restrict__ s,
                                    __bf16* __restrict__ d, int n4) {
  const int i = blockIdx.x * blockDim.x + threadIdx.x;
  if (i >= n4) return;
  float4 v = ((const float4*)s)[i];
  bf16x4 o; o[0]=(__bf16)v.x; o[1]=(__bf16)v.y; o[2]=(__bf16)v.z; o[3]=(__bf16)v.w;
  ((bf16x4*)d)[i] = o;
}

__global__ void bias_combine_kernel(const float* __restrict__ a,
                                    const float* __restrict__ b,
                                    float* __restrict__ d, int n) {
  const int i = blockIdx.x * blockDim.x + threadIdx.x;
  if (i < n) d[i] = a[i] + b[i];
}

__global__ void init_state_kernel(const float* __restrict__ eh,   // [2,B,H]
                                  const float* __restrict__ ec,   // [2,B,H]
                                  __bf16* __restrict__ h0, __bf16* __restrict__ h1,
                                  float* __restrict__ c0, float* __restrict__ c1,
                                  int* __restrict__ tok,
                                  int* __restrict__ lctr, int* __restrict__ fctr) {
  const int i = blockIdx.x * blockDim.x + threadIdx.x;
  if (i < kB * kH) {
    h0[i] = (__bf16)eh[i];
    h1[i] = (__bf16)eh[kB * kH + i];
    c0[i] = ec[i];
    c1[i] = ec[kB * kH + i];
    if (i < kB) tok[i] = 2;             // SOS
    if (i < kT) { lctr[i] = 0; fctr[i] = 0; }
  }
}

__global__ void logsoftmax_sub_kernel(float* __restrict__ out,
                                      const float* __restrict__ lse, int n4) {
  const int i = blockIdx.x * blockDim.x + threadIdx.x;
  if (i >= n4) return;
  const int row = (i * 4) / kV;          // row = b*T + t (kV divisible by 4)
  const float l = lse[row];
  float4 v = ((const float4*)out)[i];
  v.x -= l; v.y -= l; v.z -= l; v.w -= l;
  ((float4*)out)[i] = v;
}

// ---------------- host launch ----------------
extern "C" void kernel_launch(void* const* d_in, const int* in_sizes, int n_in,
                              void* d_out, int out_size, void* d_ws, size_t ws_size,
                              hipStream_t stream) {
  const float* enc_h = (const float*)d_in[1];
  const float* enc_c = (const float*)d_in[2];
  const float* emb   = (const float*)d_in[3];
  const float* wih0  = (const float*)d_in[4];
  const float* whh0  = (const float*)d_in[5];
  const float* bih0  = (const float*)d_in[6];
  const float* bhh0  = (const float*)d_in[7];
  const float* wih1  = (const float*)d_in[8];
  const float* whh1  = (const float*)d_in[9];
  const float* bih1  = (const float*)d_in[10];
  const float* bhh1  = (const float*)d_in[11];
  const float* fcw   = (const float*)d_in[12];
  const float* fcb   = (const float*)d_in[13];
  float* out = (float*)d_out;

  // workspace carve (256B aligned)
  char* base = (char*)d_ws;
  size_t off = 0;
  auto carve = [&](size_t bytes) -> void* {
    off = (off + 255) & ~(size_t)255;
    void* p = base + off;
    off += bytes;
    return p;
  };
  __bf16* wih0_b = (__bf16*)carve((size_t)4 * kH * kE * 2);
  __bf16* whh0_b = (__bf16*)carve((size_t)4 * kH * kH * 2);
  __bf16* wih1_b = (__bf16*)carve((size_t)4 * kH * kH * 2);
  __bf16* whh1_b = (__bf16*)carve((size_t)4 * kH * kH * 2);
  __bf16* fcw_b  = (__bf16*)carve((size_t)kV * kH * 2);
  float*  b0c    = (float*)carve((size_t)4 * kH * 4);
  float*  b1c    = (float*)carve((size_t)4 * kH * 4);
  __bf16* h0buf[2] = { (__bf16*)carve((size_t)kB * kH * 2), (__bf16*)carve((size_t)kB * kH * 2) };
  __bf16* h1buf[2] = { (__bf16*)carve((size_t)kB * kH * 2), (__bf16*)carve((size_t)kB * kH * 2) };
  float*  c0     = (float*)carve((size_t)kB * kH * 4);
  float*  c1     = (float*)carve((size_t)kB * kH * 4);
  int*    tok    = (int*)carve(kB * 4);
  float*  pm     = (float*)carve((size_t)kFCBlocks * kB * 4);
  int*    pa     = (int*)carve((size_t)kFCBlocks * kB * 4);
  float*  ps     = (float*)carve((size_t)kFCBlocks * kB * 4);
  float*  lse    = (float*)carve((size_t)kB * kT * 4);
  int*    lctr   = (int*)carve(kT * 4);
  int*    fctr   = (int*)carve(kT * 4);

  // fp32 -> bf16 weight conversion (per-launch; deterministic)
  auto cvt = [&](const float* s, __bf16* d, size_t n) {
    int n4 = (int)(n / 4);
    cvt_f32_bf16_kernel<<<(n4 + 255) / 256, 256, 0, stream>>>(s, d, n4);
  };
  cvt(wih0, wih0_b, (size_t)4 * kH * kE);
  cvt(whh0, whh0_b, (size_t)4 * kH * kH);
  cvt(wih1, wih1_b, (size_t)4 * kH * kH);
  cvt(whh1, whh1_b, (size_t)4 * kH * kH);
  cvt(fcw,  fcw_b,  (size_t)kV * kH);
  bias_combine_kernel<<<(4 * kH + 255) / 256, 256, 0, stream>>>(bih0, bhh0, b0c, 4 * kH);
  bias_combine_kernel<<<(4 * kH + 255) / 256, 256, 0, stream>>>(bih1, bhh1, b1c, 4 * kH);
  init_state_kernel<<<(kB * kH + 255) / 256, 256, 0, stream>>>(
      enc_h, enc_c, h0buf[0], h1buf[0], c0, c1, tok, lctr, fctr);

  for (int t = 0; t < kT; ++t) {
    const int cur = t & 1, nxt = (t + 1) & 1;
    lstm2_step_kernel<<<kLstmWGs, 256, 0, stream>>>(
        emb, tok, wih0_b, whh0_b, b0c, wih1_b, whh1_b, b1c,
        h0buf[cur], h0buf[nxt], c0, h1buf[cur], h1buf[nxt], c1, lctr + t);
    fc_logits_kernel<<<kFCBlocks, 256, 0, stream>>>(
        h1buf[nxt], fcw_b, fcb, out, t, pm, pa, ps, fctr + t, tok, lse);
  }

  // final log_softmax: out -= lse[b*T+t]
  const int n4 = kB * kT * kV / 4;
  logsoftmax_sub_kernel<<<(n4 + 255) / 256, 256, 0, stream>>>(out, lse, n4);
}
